// BitLinear_64570538328617
// MI455X (gfx1250) — 8-bit matrix operand order repaired and checked by reading, compiled, not run on hardware
//
#include <hip/hip_runtime.h>
#include <cstdint>

typedef int v8i __attribute__((ext_vector_type(8)));
typedef int vint4 __attribute__((vector_size(16)));

constexpr int M_DIM = 4 * 4096;
constexpr int K_DIM = 2048;
constexpr int N_DIM = 8192;
constexpr long long W_ELEMS = (long long)N_DIM * K_DIM;

constexpr float EPS = 1e-5f;
constexpr float NORM_EPS = 1e-6f;

#if __has_builtin(__builtin_amdgcn_global_load_async_to_lds_b128) && \
    __has_builtin(__builtin_amdgcn_s_wait_asynccnt)
#define HAS_ASYNC_LDS 1
#else
#define HAS_ASYNC_LDS 0
#endif

typedef __attribute__((address_space(1))) vint4  g_vint4;
typedef __attribute__((address_space(3))) vint4  l_vint4;

__device__ __forceinline__ void copy16_g2lds(const int8_t* gsrc, char* ldst) {
#if HAS_ASYNC_LDS
  __builtin_amdgcn_global_load_async_to_lds_b128(
      (g_vint4*)gsrc, (l_vint4*)ldst, 0, 0);
#else
  *(int4*)ldst = *(const int4*)gsrc;
#endif
}

#if HAS_ASYNC_LDS
#define ASYNC_WAIT(n) __builtin_amdgcn_s_wait_asynccnt(n)
#else
#define ASYNC_WAIT(n)
#endif


__global__ __launch_bounds__(256) void wabs_partial_kernel(
    const float* __restrict__ w, float* __restrict__ partials) {
  __shared__ float red[256];
  float s = 0.f;
  long long stride = (long long)gridDim.x * 256;
  for (long long i = (long long)blockIdx.x * 256 + threadIdx.x; i < W_ELEMS; i += stride)
    s += fabsf(w[i]);
  red[threadIdx.x] = s;
  __syncthreads();
  for (int off = 128; off > 0; off >>= 1) {
    if (threadIdx.x < (unsigned)off) red[threadIdx.x] += red[threadIdx.x + off];
    __syncthreads();
  }
  if (threadIdx.x == 0) partials[blockIdx.x] = red[0];
}

__global__ __launch_bounds__(256) void wabs_final_kernel(
    const float* __restrict__ partials, float* __restrict__ meta) {
  __shared__ float red[256];
  float s = 0.f;
#pragma unroll
  for (int i = 0; i < 4; ++i) s += partials[threadIdx.x + i * 256];
  red[threadIdx.x] = s;
  __syncthreads();
  for (int off = 128; off > 0; off >>= 1) {
    if (threadIdx.x < (unsigned)off) red[threadIdx.x] += red[threadIdx.x + off];
    __syncthreads();
  }
  if (threadIdx.x == 0) {
    float mean = red[0] / (float)W_ELEMS;
    meta[0] = red[0];
    meta[1] = mean + EPS;
  }
}

__global__ __launch_bounds__(256) void wquant_kernel(
    const float* __restrict__ w, const float* __restrict__ meta,
    int8_t* __restrict__ w_i8) {
  float scale = 1.f / meta[1];
  long long base = ((long long)blockIdx.x * 256 + threadIdx.x) * 4;
  float4 v = *(const float4*)(w + base);
  char4 q;
  q.x = (int8_t)fminf(fmaxf(rintf(v.x * scale), -1.f), 1.f);
  q.y = (int8_t)fminf(fmaxf(rintf(v.y * scale), -1.f), 1.f);
  q.z = (int8_t)fminf(fmaxf(rintf(v.z * scale), -1.f), 1.f);
  q.w = (int8_t)fminf(fmaxf(rintf(v.w * scale), -1.f), 1.f);
  *(char4*)(w_i8 + base) = q;
}

__global__ __launch_bounds__(256) void act_quant_kernel(
    const float* __restrict__ x, const float* __restrict__ g,
    int8_t* __restrict__ a_i8, float* __restrict__ inv_sa) {
  __shared__ float red[256];
  const int row = blockIdx.x;
  const float* xr = x + (long long)row * K_DIM;

  float v[8];
  float ss = 0.f;
#pragma unroll
  for (int i = 0; i < 8; ++i) {
    v[i] = xr[threadIdx.x + i * 256];
    ss += v[i] * v[i];
  }
  red[threadIdx.x] = ss;
  __syncthreads();
  for (int off = 128; off > 0; off >>= 1) {
    if (threadIdx.x < (unsigned)off) red[threadIdx.x] += red[threadIdx.x + off];
    __syncthreads();
  }
  float rms = rsqrtf(red[0] / (float)K_DIM + NORM_EPS);
  __syncthreads();

  float xn[8];
  float amax = 0.f;
#pragma unroll
  for (int i = 0; i < 8; ++i) {
    xn[i] = v[i] * rms * g[threadIdx.x + i * 256];
    amax = fmaxf(amax, fabsf(xn[i]));
  }
  red[threadIdx.x] = amax;
  __syncthreads();
  for (int off = 128; off > 0; off >>= 1) {
    if (threadIdx.x < (unsigned)off)
      red[threadIdx.x] = fmaxf(red[threadIdx.x], red[threadIdx.x + off]);
    __syncthreads();
  }
  float am = red[0] + EPS;
  float scale = 127.f / am;
  if (threadIdx.x == 0) inv_sa[row] = am / 127.f;

  int8_t* ar = a_i8 + (long long)row * K_DIM;
#pragma unroll
  for (int i = 0; i < 8; ++i) {
    float q = fminf(fmaxf(rintf(xn[i] * scale), -128.f), 127.f);
    ar[threadIdx.x + i * 256] = (int8_t)q;
  }
}

constexpr int BM = 128, BN = 256, BK = 128;
constexpr int LDS_STRIDE = 144;
constexpr int A_TILE_BYTES = BM * LDS_STRIDE;
constexpr int B_TILE_BYTES = BN * LDS_STRIDE;
constexpr int STAGE_BYTES = A_TILE_BYTES + B_TILE_BYTES;
constexpr int SMEM_BYTES = 2 * STAGE_BYTES;
constexpr int COPIES_PER_STAGE = 12;

__device__ __forceinline__ void load_stage(
    const int8_t* __restrict__ A, const int8_t* __restrict__ W,
    int m_blk, int n_blk, int k0, char* sA, char* sB, int tid) {
#pragma unroll
  for (int i = 0; i < 4; ++i) {
    int c = tid + i * 256;
    int row = c >> 3, ko = (c & 7) * 16;
    copy16_g2lds(A + (long long)(m_blk + row) * K_DIM + k0 + ko,
                 sA + row * LDS_STRIDE + ko);
  }
#pragma unroll
  for (int i = 0; i < 8; ++i) {
    int c = tid + i * 256;
    int row = c >> 3, ko = (c & 7) * 16;
    copy16_g2lds(W + (long long)(n_blk + row) * K_DIM + k0 + ko,
                 sB + row * LDS_STRIDE + ko);
  }
}

__global__ __launch_bounds__(256) void gemm_iu8_kernel(
    const int8_t* __restrict__ A, const int8_t* __restrict__ W,
    const float* __restrict__ inv_sa, const float* __restrict__ meta,
    float* __restrict__ out) {
  extern __shared__ char smem[];

  const int tid  = threadIdx.x;
  const int lane = tid & 31;
  const int wave = tid >> 5;
  const int half = lane >> 4;
  const int l16  = lane & 15;
  const int wr = wave & 1;
  const int wc = wave >> 1;
  const int m_blk = blockIdx.y * BM;
  const int n_blk = blockIdx.x * BN;

  v8i acc[4][4] = {};

  load_stage(A, W, m_blk, n_blk, 0, smem, smem + A_TILE_BYTES, tid);

  for (int kt = 0; kt < K_DIM / BK; ++kt) {
    if (kt + 1 < K_DIM / BK) {
      char* nA = smem + ((kt + 1) & 1) * STAGE_BYTES;
      load_stage(A, W, m_blk, n_blk, (kt + 1) * BK, nA, nA + A_TILE_BYTES, tid);
      ASYNC_WAIT(COPIES_PER_STAGE);
    } else {
      ASYNC_WAIT(0);
    }
    __syncthreads();

    const char* cA = smem + (kt & 1) * STAGE_BYTES;
    const char* cB = cA + A_TILE_BYTES;
#pragma unroll
    for (int kc = 0; kc < 2; ++kc) {
      const int kb = kc * 64;
      v8i a[4];
#pragma unroll
      for (int ti = 0; ti < 4; ++ti) {
        const char* p0 = cA + (wr * 64 + ti * 16 + l16) * LDS_STRIDE + kb + half * 8;
#pragma unroll
        for (int p = 0; p < 4; ++p) {
          int2 d = *(const int2*)(p0 + p * 16);
          a[ti][2 * p]     = d.x;
          a[ti][2 * p + 1] = d.y;
        }
      }
      v8i b[4];
#pragma unroll
      for (int tj = 0; tj < 4; ++tj) {
        const char* p0 = cB + (wc * 64 + tj * 16 + l16) * LDS_STRIDE + kb + half * 8;
#pragma unroll
        for (int p = 0; p < 4; ++p) {
          int2 d = *(const int2*)(p0 + p * 16);
          b[tj][2 * p]     = d.x;
          b[tj][2 * p + 1] = d.y;
        }
      }
#pragma unroll
      for (int ti = 0; ti < 4; ++ti)
#pragma unroll
        for (int tj = 0; tj < 4; ++tj)
          acc[ti][tj] = __builtin_amdgcn_wmma_i32_16x16x64_iu8(
               true, a[ti],  true, b[tj], acc[ti][tj],
               false,  false);
    }
    __syncthreads();
  }

  const float inv_sw = meta[1];
  const int m_wave = m_blk + wr * 64;
  float sa[4][8];
#pragma unroll
  for (int ti = 0; ti < 4; ++ti)
#pragma unroll
    for (int r = 0; r < 8; ++r)
      sa[ti][r] = inv_sa[m_wave + ti * 16 + r + 8 * half] * inv_sw;

#pragma unroll
  for (int ti = 0; ti < 4; ++ti) {
#pragma unroll
    for (int tj = 0; tj < 4; ++tj) {
      const int n = n_blk + wc * 64 + tj * 16 + l16;
#pragma unroll
      for (int r = 0; r < 8; ++r) {
        const int m = m_wave + ti * 16 + r + 8 * half;
        out[(long long)m * N_DIM + n] = (float)acc[ti][tj][r] * sa[ti][r];
      }
    }
  }
}

extern "C" void kernel_launch(void* const* d_in, const int* in_sizes, int n_in,
                              void* d_out, int out_size, void* d_ws, size_t ws_size,
                              hipStream_t stream) {
  const float* x  = (const float*)d_in[0];
  const float* g  = (const float*)d_in[1];
  const float* w  = (const float*)d_in[2];
  float* out = (float*)d_out;

  char* ws = (char*)d_ws;
  float*  meta     = (float*)ws;
  float*  partials = (float*)(ws + 256);
  float*  inv_sa   = (float*)(ws + 8192);
  int8_t* a_i8     = (int8_t*)(ws + 131072);
  int8_t* w_i8     = (int8_t*)(ws + 131072 + (size_t)M_DIM * K_DIM);

  (void)hipMemsetAsync(d_ws, 0, 256, stream);

  wabs_partial_kernel<<<1024, 256, 0, stream>>>(w, partials);
  wabs_final_kernel<<<1, 256, 0, stream>>>(partials, meta);
  wquant_kernel<<<(int)(W_ELEMS / (256 * 4)), 256, 0, stream>>>(w, meta, w_i8);
  act_quant_kernel<<<M_DIM, 256, 0, stream>>>(x, g, a_i8, inv_sa);

  dim3 grid(N_DIM / BN, M_DIM / BM);
  gemm_iu8_kernel<<<grid, 256, SMEM_BYTES, stream>>>(a_i8, w_i8, inv_sa, meta, out);
}
